// LastEncoderLayer_69441031242068
// MI455X (gfx1250) — matrix operand order repaired and checked by reading, not compiled, not run on hardware
//
#include <hip/hip_runtime.h>
#include <hip/hip_bf16.h>
#include <cstdint>
#include <cstddef>

// ---------------------------------------------------------------------------
// Transformer encoder layer for MI455X (gfx1250), bf16 WMMA + f32 accumulate.
// B=16, S=1024, E=1024, H=16, hd=64, FFN=4096.
// Tile staging uses gfx1250 async global->LDS DMA (ASYNCcnt).
// ---------------------------------------------------------------------------

typedef unsigned short u16;
typedef __attribute__((ext_vector_type(8)))  u16    us8;     // 16 bytes
typedef __attribute__((ext_vector_type(16))) u16    us16;    // 32 bytes
typedef __attribute__((ext_vector_type(16))) __bf16 bf16x16; // WMMA A/B operand
typedef __attribute__((ext_vector_type(8)))  float  f32x8;   // WMMA C/D operand

#define B_   16
#define S_   1024
#define E_   1024
#define H_   16
#define HD_  64
#define FFN_ 4096
#define BS_  (B_ * S_)

__device__ __forceinline__ u16 f2bf(float f) {
  unsigned u = __builtin_bit_cast(unsigned, f);
  u += 0x7fffu + ((u >> 16) & 1u);   // round-to-nearest-even
  return (u16)(u >> 16);
}

__device__ __forceinline__ bf16x16 join16(us8 lo, us8 hi) {
  us16 v = __builtin_shufflevector(lo, hi, 0, 1, 2, 3, 4, 5, 6, 7,
                                   8, 9, 10, 11, 12, 13, 14, 15);
  return __builtin_bit_cast(bf16x16, v);
}
__device__ __forceinline__ bf16x16 as_bf(us16 v) {
  return __builtin_bit_cast(bf16x16, v);
}

// --- gfx1250 async global->LDS DMA (16B per lane), tracked by ASYNCcnt -----
__device__ __forceinline__ void async_b128(void* lds_ptr, const void* gptr) {
  unsigned lds_off = (unsigned)(uintptr_t)lds_ptr;  // low 32b of LDS aperture
  asm volatile("global_load_async_to_lds_b128 %0, %1, off"
               :: "v"(lds_off), "v"(gptr)
               : "memory");
}
__device__ __forceinline__ void wait_async0() {
  asm volatile("s_wait_asynccnt 0x0" ::: "memory");
}

// --------------------------- fp32 -> bf16 convert ---------------------------
__global__ void cvt_bf16_kernel(const float* __restrict__ in,
                                u16* __restrict__ out, int n) {
  for (int i = blockIdx.x * blockDim.x + threadIdx.x; i < n;
       i += gridDim.x * blockDim.x)
    out[i] = f2bf(in[i]);
}

// ------------------------- layernorm -> bf16 row ----------------------------
__global__ __launch_bounds__(256) void ln_bf16_kernel(
    const float* __restrict__ x, const float* __restrict__ g,
    const float* __restrict__ b, u16* __restrict__ out) {
  const int row = blockIdx.x;
  const float* xr = x + (size_t)row * E_;
  const int tid = threadIdx.x, lane = tid & 31, w = tid >> 5;
  float s = 0.f, q = 0.f;
  for (int i = tid; i < E_; i += 256) { float v = xr[i]; s += v; q += v * v; }
#pragma unroll
  for (int off = 16; off > 0; off >>= 1) {
    s += __shfl_xor(s, off, 32);
    q += __shfl_xor(q, off, 32);
  }
  __shared__ float rs[8], rq[8];
  if (lane == 0) { rs[w] = s; rq[w] = q; }
  __syncthreads();
  float ts = 0.f, tq = 0.f;
#pragma unroll
  for (int j = 0; j < 8; ++j) { ts += rs[j]; tq += rq[j]; }
  const float mu = ts * (1.0f / E_);
  const float var = tq * (1.0f / E_) - mu * mu;
  const float inv = rsqrtf(var + 1e-6f);
  for (int i = tid; i < E_; i += 256) {
    float v = (xr[i] - mu) * inv * g[i] + b[i];
    out[(size_t)row * E_ + i] = f2bf(v);
  }
}

// ------------------------------- WMMA GEMM ----------------------------------
// C[M,N] = A[M,K](bf16) * Bw[N,K](bf16)^T + bias, templated epilogue.
#define EPI_BF16      0  // out bf16 = c + bias
#define EPI_F32_RESID 1  // out f32  = c + bias + resid
#define EPI_BF16_GELU 2  // out bf16 = gelu(c + bias)

template <int EPI>
__global__ __launch_bounds__(256) void gemm_bf16_nt(
    const u16* __restrict__ A, const u16* __restrict__ Bw,
    const float* __restrict__ bias, const float* __restrict__ resid,
    float* __restrict__ outF, u16* __restrict__ outB, int M, int N, int K) {
  (void)M;
  __shared__ __align__(32) u16 sA[2][128][40];  // 128x32 tile, pad 8
  __shared__ __align__(32) u16 sB[2][64][48];   // 64x32 tile, pad 16

  const int tid = threadIdx.x, lane = tid & 31, w = tid >> 5;
  const int g = lane >> 4, ln = lane & 15;
  const int wm = (w & 3) * 32, wn = (w >> 2) * 32;
  const int rowB = blockIdx.y * 128, colB = blockIdx.x * 64;
  const int nk = K >> 5;

  __builtin_prefetch(bias + colB + wn, 0, 0);

  f32x8 acc[2][2] = {};

  // async DMA one (A,B) K-tile pair into LDS buffer `buf`
  auto stage = [&](int buf, int kt) {
    const int k0 = kt << 5;
#pragma unroll
    for (int i = 0; i < 2; ++i) {
      int c = tid + i * 256, r = c >> 2, kc = (c & 3) * 8;
      async_b128(&sA[buf][r][kc], A + (size_t)(rowB + r) * K + k0 + kc);
    }
    {
      int r = tid >> 2, kc = (tid & 3) * 8;
      async_b128(&sB[buf][r][kc], Bw + (size_t)(colB + r) * K + k0 + kc);
    }
  };
  auto compute = [&](int buf) {
    bf16x16 af[2], bfr[2];
#pragma unroll
    for (int rt = 0; rt < 2; ++rt) {
      const u16* p = &sA[buf][wm + rt * 16 + ln][0];
      af[rt] = join16(*(const us8*)(p + g * 8), *(const us8*)(p + 16 + g * 8));
    }
#pragma unroll
    for (int ct = 0; ct < 2; ++ct) {
      const u16* p = &sB[buf][wn + ct * 16 + ln][0];
      bfr[ct] = join16(*(const us8*)(p + g * 8), *(const us8*)(p + 16 + g * 8));
    }
#pragma unroll
    for (int rt = 0; rt < 2; ++rt)
#pragma unroll
      for (int ct = 0; ct < 2; ++ct)
        acc[rt][ct] = __builtin_amdgcn_wmma_f32_16x16x32_bf16(
            false, af[rt], false, bfr[ct], (short)0, acc[rt][ct], false, false);
  };

  stage(0, 0);
  wait_async0();
  __syncthreads();
  for (int kt = 0; kt < nk; ++kt) {
    const int buf = kt & 1;
    if (kt + 1 < nk) stage(buf ^ 1, kt + 1);  // DMA overlaps the WMMAs
    compute(buf);
    if (kt + 1 < nk) wait_async0();
    __syncthreads();
  }

#pragma unroll
  for (int rt = 0; rt < 2; ++rt)
#pragma unroll
    for (int ct = 0; ct < 2; ++ct) {
      const int col = colB + wn + ct * 16 + ln;
      const float bs = bias[col];
#pragma unroll
      for (int v = 0; v < 8; ++v) {
        const int row = rowB + wm + rt * 16 + v + 8 * g;
        float val = acc[rt][ct][v] + bs;
        const size_t idx = (size_t)row * N + col;
        if constexpr (EPI == EPI_F32_RESID) {
          outF[idx] = val + resid[idx];
        } else if constexpr (EPI == EPI_BF16_GELU) {
          float ge = 0.5f * val * (1.0f + erff(val * 0.70710678118f));
          outB[idx] = f2bf(ge);
        } else {
          outB[idx] = f2bf(val);
        }
      }
    }
}

// ------------------------------- attention ----------------------------------
// One block per (b, h, 32 query rows). Scores row-block kept in LDS (128 KB),
// exploiting the 320 KB WGP LDS. WMMA for Q*K^T and P*V.
__global__ __launch_bounds__(256) void attn_kernel(
    const u16* __restrict__ qkv,  // [BS, 3E] bf16
    u16* __restrict__ attnO) {    // [BS, E]  bf16
  extern __shared__ char smem[];
  float* sS = (float*)smem;                                // [32][S_] fp32
  u16* sQ = (u16*)(smem + 32 * S_ * sizeof(float));        // [32][72]
  u16* sKV = sQ + 32 * 72;                                 // [64][72] / Vt [64][48]

  const int tid = threadIdx.x, lane = tid & 31, w = tid >> 5;
  const int g = lane >> 4, ln = lane & 15;

  const int bid = blockIdx.x;
  const int qt = bid & (S_ / 32 - 1);
  const int h = (bid / (S_ / 32)) & (H_ - 1);
  const int b = bid / ((S_ / 32) * H_);
  const int ld = 3 * E_;
  const size_t rowQ = (size_t)(b * S_ + qt * 32);
  const int qcol = h * HD_, kcol = E_ + h * HD_, vcol = 2 * E_ + h * HD_;

  // stage Q tile [32 x 64] via async DMA
  {
    int r = tid >> 3, kc = (tid & 7) * 8;
    async_b128(&sQ[r * 72 + kc], qkv + (rowQ + r) * ld + qcol + kc);
  }
  wait_async0();
  __syncthreads();

  // per-wave Q fragments (2 K-steps over d=64)
  bf16x16 aq[2];
  {
    const u16* p = &sQ[((w & 1) * 16 + ln) * 72];
#pragma unroll
    for (int s = 0; s < 2; ++s)
      aq[s] = join16(*(const us8*)(p + s * 32 + g * 8),
                     *(const us8*)(p + s * 32 + 16 + g * 8));
  }

  // scores = (Q K^T) / 8, streamed over key tiles of 64 (async K staging)
  for (int kt = 0; kt < S_ / 64; ++kt) {
#pragma unroll
    for (int i = 0; i < 2; ++i) {
      int c = tid + i * 256, r = c >> 3, kc = (c & 7) * 8;
      async_b128(&sKV[r * 72 + kc],
                 qkv + (size_t)(b * S_ + kt * 64 + r) * ld + kcol + kc);
    }
    wait_async0();
    __syncthreads();
    f32x8 acc = {};
    const u16* p = &sKV[((w >> 1) * 16 + ln) * 72];
#pragma unroll
    for (int s = 0; s < 2; ++s)
      acc = __builtin_amdgcn_wmma_f32_16x16x32_bf16(
          false, aq[s], false,
          join16(*(const us8*)(p + s * 32 + g * 8),
                 *(const us8*)(p + s * 32 + 16 + g * 8)),
          (short)0, acc, false, false);
    const int colb = kt * 64 + (w >> 1) * 16 + ln;
#pragma unroll
    for (int v = 0; v < 8; ++v) {
      int m = (w & 1) * 16 + v + 8 * g;
      sS[m * S_ + colb] = acc[v] * 0.125f;  // 1/sqrt(64)
    }
    __syncthreads();
  }

  // softmax: 8 lanes per row (lane groups are shfl-aligned)
  {
    int r = tid >> 3, base = (tid & 7) * (S_ / 8);
    float* rowp = sS + r * S_ + base;
    float mx = -3.4e38f;
    for (int j = 0; j < S_ / 8; ++j) mx = fmaxf(mx, rowp[j]);
#pragma unroll
    for (int off = 1; off < 8; off <<= 1) mx = fmaxf(mx, __shfl_xor(mx, off, 32));
    float sum = 0.f;
    for (int j = 0; j < S_ / 8; ++j) {
      float e = __expf(rowp[j] - mx);
      rowp[j] = e;
      sum += e;
    }
#pragma unroll
    for (int off = 1; off < 8; off <<= 1) sum += __shfl_xor(sum, off, 32);
    const float inv = 1.0f / sum;
    for (int j = 0; j < S_ / 8; ++j) rowp[j] *= inv;
  }
  __syncthreads();

  // out = P * V, streamed over kv tiles of 32; V staged transposed [d][kv]
  f32x8 accO = {};
  for (int vt = 0; vt < S_ / 32; ++vt) {
    {
      int kvr = tid >> 3, dc = (tid & 7) * 8;
      us8 vv = *(const us8*)(qkv + (size_t)(b * S_ + vt * 32 + kvr) * ld +
                             vcol + dc);
#pragma unroll
      for (int j = 0; j < 8; ++j) sKV[(dc + j) * 48 + kvr] = vv[j];
    }
    __syncthreads();
    const float* pr = sS + ((w & 1) * 16 + ln) * S_ + vt * 32;
    us16 t = {};
#pragma unroll
    for (int j = 0; j < 8; ++j) {
      t[j] = f2bf(pr[g * 8 + j]);
      t[j + 8] = f2bf(pr[16 + g * 8 + j]);
    }
    const u16* p = &sKV[((w >> 1) * 16 + ln) * 48];
    accO = __builtin_amdgcn_wmma_f32_16x16x32_bf16(
        false, as_bf(t), false,
        join16(*(const us8*)(p + g * 8), *(const us8*)(p + 16 + g * 8)),
        (short)0, accO, false, false);
    __syncthreads();
  }

  {
    const int gcol = h * HD_ + (w >> 1) * 16 + ln;
#pragma unroll
    for (int v = 0; v < 8; ++v) {
      size_t grow = rowQ + (w & 1) * 16 + v + 8 * g;
      attnO[grow * E_ + gcol] = f2bf(accO[v]);
    }
  }
}

// --------------------------------- launcher ---------------------------------
extern "C" void kernel_launch(void* const* d_in, const int* in_sizes, int n_in,
                              void* d_out, int out_size, void* d_ws,
                              size_t ws_size, hipStream_t stream) {
  (void)in_sizes; (void)n_in; (void)out_size; (void)ws_size;
  const float* x    = (const float*)d_in[0];
  const float* ln1w = (const float*)d_in[1];
  const float* ln1b = (const float*)d_in[2];
  const float* ln2w = (const float*)d_in[3];
  const float* ln2b = (const float*)d_in[4];
  const float* ipw  = (const float*)d_in[5];
  const float* ipb  = (const float*)d_in[6];
  const float* outw = (const float*)d_in[7];
  const float* outb = (const float*)d_in[8];
  const float* w1   = (const float*)d_in[9];
  const float* b1   = (const float*)d_in[10];
  const float* w2   = (const float*)d_in[11];
  const float* b2   = (const float*)d_in[12];

  char* ws = (char*)d_ws;
  u16* WQKV  = (u16*)(ws + 0);          // 3E*E bf16      (6.0 MiB)
  u16* WOUT  = (u16*)(ws + 6291456);    // E*E bf16       (2.0 MiB)
  u16* W1b   = (u16*)(ws + 8388608);    // FFN*E bf16     (8.0 MiB)
  u16* W2b   = (u16*)(ws + 16777216);   // E*FFN bf16     (8.0 MiB)
  u16* HBF   = (u16*)(ws + 25165824);   // [BS,E] bf16    (32 MiB, reused)
  float* X1  = (float*)(ws + 58720256); // [BS,E] f32     (64 MiB)
  u16* QKV   = (u16*)(ws + 125829120);  // [BS,3E] bf16   (96 MiB)
  u16* ATTN  = (u16*)(ws + 226492416);  // [BS,E] bf16    (32 MiB)
  u16* FFN1a = QKV;                     // [BS,FFN] bf16 overlays QKV+ATTN

  // weights -> bf16
  cvt_bf16_kernel<<<4096, 256, 0, stream>>>(ipw, WQKV, 3 * E_ * E_);
  cvt_bf16_kernel<<<2048, 256, 0, stream>>>(outw, WOUT, E_ * E_);
  cvt_bf16_kernel<<<4096, 256, 0, stream>>>(w1, W1b, FFN_ * E_);
  cvt_bf16_kernel<<<4096, 256, 0, stream>>>(w2, W2b, E_ * FFN_);

  // h = LN1(x) -> bf16
  ln_bf16_kernel<<<BS_, 256, 0, stream>>>(x, ln1w, ln1b, HBF);

  // qkv = h @ Wqkv^T + b (bf16 out)
  gemm_bf16_nt<EPI_BF16><<<dim3(3 * E_ / 64, BS_ / 128), 256, 0, stream>>>(
      HBF, WQKV, ipb, nullptr, nullptr, QKV, BS_, 3 * E_, E_);

  // attention
  const size_t smem = 32 * S_ * sizeof(float) + 32 * 72 * 2 + 64 * 72 * 2;
  hipFuncSetAttribute((const void*)attn_kernel,
                      hipFuncAttributeMaxDynamicSharedMemorySize, (int)smem);
  attn_kernel<<<B_ * H_ * (S_ / 32), 256, smem, stream>>>(QKV, ATTN);

  // x1 = x + attn @ Wout^T + b (f32 out)
  gemm_bf16_nt<EPI_F32_RESID><<<dim3(E_ / 64, BS_ / 128), 256, 0, stream>>>(
      ATTN, WOUT, outb, x, X1, nullptr, BS_, E_, E_);

  // h2 = LN2(x1) -> bf16
  ln_bf16_kernel<<<BS_, 256, 0, stream>>>(X1, ln2w, ln2b, HBF);

  // f1 = gelu(h2 @ W1^T + b1) (bf16 out)
  gemm_bf16_nt<EPI_BF16_GELU><<<dim3(FFN_ / 64, BS_ / 128), 256, 0, stream>>>(
      HBF, W1b, b1, nullptr, nullptr, FFN1a, BS_, FFN_, E_);

  // out = x1 + f1 @ W2^T + b2 (f32 out)
  gemm_bf16_nt<EPI_F32_RESID><<<dim3(E_ / 64, BS_ / 128), 256, 0, stream>>>(
      FFN1a, W2b, b2, X1, (float*)d_out, nullptr, BS_, E_, FFN_);
}
